// QueryGenerator_58102317580863
// MI455X (gfx1250) — matrix operand order repaired and checked by reading, not compiled, not run on hardware
//
#include <hip/hip_runtime.h>
#include <hip/hip_bf16.h>

// ---------------- problem constants (match reference) ----------------
#define BATCH 2
#define HH    1024
#define WW    1024
#define NNZ   100000
#define CC    16
#define CIN   128
#define CHID  128
#define THR   0.95f
#define QMAX  (NNZ * (CC - 1))   // 1,500,000
#define GRIDN (BATCH * HH * WW)  // 2,097,152
#define BM    64                 // M-rows per workgroup

typedef __attribute__((ext_vector_type(16))) _Float16 v16h;
typedef __attribute__((ext_vector_type(8)))  float    v8f;
typedef __attribute__((ext_vector_type(4)))  float    v4f;
typedef __attribute__((ext_vector_type(4)))  unsigned int v4u;

union Frag16 { v16h h; v4u q[2]; };

// ---- gfx1250 async global->LDS copy (16 bytes), SADDR form ----------
__device__ __forceinline__ void async_copy_b128(void* lds_ptr,
                                                unsigned long long gbase,
                                                unsigned byte_off) {
    unsigned lds = (unsigned)(unsigned long long)lds_ptr;  // low 32 = LDS addr
    asm volatile("global_load_async_to_lds_b128 %0, %1, %2 offset:0"
                 :: "v"(lds), "v"(byte_off), "s"(gbase) : "memory");
}
__device__ __forceinline__ void wait_asynccnt0() {
    asm volatile("s_wait_asynccnt 0x0" ::: "memory");
}

// ---------------------------------------------------------------------
// Kernel 1: per-site softmax -> CDF -> min count with cdf > THR
// ---------------------------------------------------------------------
__global__ __launch_bounds__(256) void k_counts(const float* __restrict__ occ,
                                                int* __restrict__ counts) {
    int n = blockIdx.x * blockDim.x + threadIdx.x;
    if (n >= NNZ) return;
    float v[CC];
    float m = -1e30f;
#pragma unroll
    for (int c = 0; c < CC; ++c) { v[c] = occ[n * CC + c]; m = fmaxf(m, v[c]); }
    float s = 0.f;
#pragma unroll
    for (int c = 0; c < CC; ++c) { v[c] = __expf(v[c] - m); s += v[c]; }
    float thr = THR * s;
    float cdf = 0.f;
    int cnt = CC - 1;
    bool found = false;
#pragma unroll
    for (int c = 0; c < CC; ++c) {
        cdf += v[c];
        if (!found && cdf > thr) { cnt = c; found = true; }
    }
    counts[n] = cnt;
}

// ---------------------------------------------------------------------
// Kernel 2: single-block inclusive scan of counts -> cum
// ---------------------------------------------------------------------
__global__ __launch_bounds__(1024) void k_scan(const int* __restrict__ counts,
                                               int* __restrict__ cum) {
    __shared__ int tmp[1024];
    __shared__ int carry;
    int t = threadIdx.x;
    if (t == 0) carry = 0;
    __syncthreads();
    for (int base = 0; base < NNZ; base += 1024) {
        int idx = base + t;
        int val = (idx < NNZ) ? counts[idx] : 0;
        tmp[t] = val;
        __syncthreads();
        for (int off = 1; off < 1024; off <<= 1) {
            int add = (t >= off) ? tmp[t - off] : 0;
            __syncthreads();
            tmp[t] += add;
            __syncthreads();
        }
        if (idx < NNZ) cum[idx] = carry + tmp[t];
        __syncthreads();
        if (t == 0) carry += tmp[1023];
        __syncthreads();
    }
}

// ---------------------------------------------------------------------
// Kernel 3: dense hash grid  (b,i,j) -> active row
// ---------------------------------------------------------------------
__global__ __launch_bounds__(256) void k_grid_init(int* __restrict__ grid) {
    int i = blockIdx.x * blockDim.x + threadIdx.x;
    if (i < GRIDN) grid[i] = -1;
}

__global__ __launch_bounds__(256) void k_grid_scatter(const int* __restrict__ occ_idx,
                                                      int* __restrict__ grid) {
    int n = blockIdx.x * blockDim.x + threadIdx.x;
    if (n >= NNZ) return;
    int b_ = occ_idx[n];
    int i_ = occ_idx[NNZ + n];
    int j_ = occ_idx[2 * NNZ + n];
    grid[(b_ * HH + i_) * WW + j_] = n;
}

// ---------------------------------------------------------------------
// Prologue: convert w1 fp32 [tap][k][n] -> fp16 transposed [tap][n][k]
// ---------------------------------------------------------------------
__global__ __launch_bounds__(256) void k_w1_to_f16(const float* __restrict__ w1,
                                                   _Float16* __restrict__ w1h) {
    int idx = blockIdx.x * blockDim.x + threadIdx.x;
    if (idx >= 9 * CIN * CHID) return;
    int tap = idx / (CIN * CHID);
    int r   = idx - tap * (CIN * CHID);
    int k   = r >> 7;    // cin
    int n   = r & 127;   // ch
    w1h[tap * CIN * CHID + n * CIN + k] = (_Float16)w1[idx];
}

// ---------------------------------------------------------------------
// Prologue: convert pixel_values fp32 -> fp16 (halves gather traffic)
// ---------------------------------------------------------------------
__global__ __launch_bounds__(256) void k_pix_to_f16(const float* __restrict__ pixel,
                                                    _Float16* __restrict__ pixel_h) {
    int idx4 = blockIdx.x * blockDim.x + threadIdx.x;
    if (idx4 >= (NNZ * CIN) / 4) return;
    v4f f = *(const v4f*)(pixel + (size_t)idx4 * 4);
    size_t o = (size_t)idx4 * 4;
    pixel_h[o + 0] = (_Float16)f.x;
    pixel_h[o + 1] = (_Float16)f.y;
    pixel_h[o + 2] = (_Float16)f.z;
    pixel_h[o + 3] = (_Float16)f.w;
}

// ---------------------------------------------------------------------
// Kernel 4: SubMConv2d 3x3 (CIN->CH) + ReLU + (CH->4) + exp  via WMMA
//   1 workgroup = 64 active sites, 8 waves; each wave owns an N-tile of
//   16 and 4 M-subtiles (4 accumulators). Staging is async global->LDS.
//   Fragment loads are batched per K-chunk so the scheduler can overlap
//   ds_load latency with back-to-back WMMAs.
// ---------------------------------------------------------------------
__global__ __launch_bounds__(256) void k_subm_conv(
        const int*      __restrict__ occ_idx,
        const _Float16* __restrict__ pixel_h,   // [NNZ][128] fp16
        const _Float16* __restrict__ w1h,       // [9][n=128][k=128] fp16
        const float*    __restrict__ b1,
        const float*    __restrict__ w2,
        const float*    __restrict__ b2,
        const int*      __restrict__ grid,
        float*          __restrict__ params) {
    __shared__ _Float16 sA[BM * CIN];        // [m][k]  16 KB
    __shared__ _Float16 sB[CHID * CIN];      // [n][k]  32 KB
    __shared__ float    sH[BM * CHID];       // [m][n]  32 KB
    __shared__ int      sIdx[BM];

    const int base = blockIdx.x * BM;
    const int tid  = threadIdx.x;
    const int lane = tid & 31;
    const int wave = tid >> 5;

    const unsigned long long pixBase = (unsigned long long)pixel_h;
    const unsigned long long wBase   = (unsigned long long)w1h;

    v8f acc[4];
#pragma unroll
    for (int m = 0; m < 4; ++m) acc[m] = (v8f){};

    for (int tap = 0; tap < 9; ++tap) {
        const int di = tap / 3 - 1;
        const int dj = tap % 3 - 1;

        __syncthreads();   // previous tap's WMMA reads of sA/sB are done

        // --- neighbor gather indices for the 64 rows of this tile ---
        if (tid < BM) {
            int site = base + tid;
            int g = -1;
            if (site < NNZ) {
                int b_ = occ_idx[site];
                int i_ = occ_idx[NNZ + site];
                int j_ = occ_idx[2 * NNZ + site];
                int ni = i_ + di, nj = j_ + dj;
                if (ni >= 0 && ni < HH && nj >= 0 && nj < WW)
                    g = grid[(b_ * HH + ni) * WW + nj];
            }
            sIdx[tid] = g;
        }

        // --- async-stage w1h[tap] (pre-transposed fp16) into sB ---
        // 32768 bytes = 2048 x b128 chunks; 8 per thread.
        {
            unsigned tapOff = (unsigned)tap * CIN * CHID * 2u;
#pragma unroll
            for (int it = 0; it < 8; ++it) {
                int c = tid + it * 256;
                async_copy_b128(&sB[c * 8], wBase, tapOff + (unsigned)c * 16u);
            }
        }
        __syncthreads();   // sIdx visible before A gather

        // --- stage gathered activations (fp16) into sA ---
        // 64 rows x 128 f16 = 1024 x b128 chunks; 4 per thread.
#pragma unroll
        for (int it = 0; it < 4; ++it) {
            int c   = tid + it * 256;
            int row = c >> 4;            // 16 chunks per row
            int seg = c & 15;
            int g   = sIdx[row];
            if (g >= 0) {
                async_copy_b128(&sA[c * 8], pixBase,
                                (unsigned)g * (CIN * 2u) + (unsigned)seg * 16u);
            } else {
                *(v4u*)(&sA[c * 8]) = (v4u){0u, 0u, 0u, 0u};
            }
        }
        wait_asynccnt0();
        __syncthreads();

        // --- WMMA: per wave, 4 M-subtiles x (K=128 in 4 chunks of 32) ---
        const int nb   = wave * 16;
        const int mrow = lane & 15;
        const int aoff = (lane < 16) ? 0 : 8;    // ISA 16-bit A layout
        const int boff = (lane < 16) ? 0 : 8;    // ISA 16-bit B layout
#pragma unroll
        for (int kb = 0; kb < CIN; kb += 32) {
            // batch all fragment loads for this K-chunk first (10 x b128)
            Frag16 bfrag;
            bfrag.q[0] = *(const v4u*)(sB + (nb + mrow) * CIN + kb + boff);
            bfrag.q[1] = *(const v4u*)(sB + (nb + mrow) * CIN + kb + boff + 16);
            Frag16 afrag[4];
#pragma unroll
            for (int m = 0; m < 4; ++m) {
                const int arow = m * 16 + mrow;
                afrag[m].q[0] = *(const v4u*)(sA + arow * CIN + kb + aoff);
                afrag[m].q[1] = *(const v4u*)(sA + arow * CIN + kb + aoff + 16);
            }
            // then 4 back-to-back WMMAs
#pragma unroll
            for (int m = 0; m < 4; ++m) {
                acc[m] = __builtin_amdgcn_wmma_f32_16x16x32_f16(
                    false, afrag[m].h, false, bfrag.h, (short)0, acc[m],
                    false, false);
            }
        }
    }

    __syncthreads();

    // --- epilogue: + b1, ReLU, stage h in LDS ---
    const int ncol  = wave * 16 + (lane & 15);
    const int mbase = (lane < 16) ? 0 : 8;
    const float bias = b1[ncol];
#pragma unroll
    for (int m = 0; m < 4; ++m)
#pragma unroll
        for (int r = 0; r < 8; ++r)
            sH[(m * 16 + mbase + r) * CHID + ncol] = fmaxf(acc[m][r] + bias, 0.f);
    __syncthreads();

    // --- tiny GEMM: h[64,128] @ w2[128,4] + b2, then exp -> params ---
    {
        int r   = tid >> 2;   // 0..63
        int col = tid & 3;
        int site = base + r;
        if (site < NNZ) {
            float s = b2[col];
#pragma unroll 8
            for (int k = 0; k < CHID; ++k)
                s += sH[r * CHID + k] * w2[k * 4 + col];
            params[(size_t)site * 4 + col] = __expf(s);
        }
    }
}

// ---------------------------------------------------------------------
// Kernel 5: ragged query expansion + Beta sampling
// ---------------------------------------------------------------------
__device__ __forceinline__ unsigned int pcg_hash(unsigned int x) {
    x = x * 747796405u + 2891336453u;
    unsigned int w = ((x >> ((x >> 28) + 4u)) ^ x) * 277803737u;
    return (w >> 22) ^ w;
}

__device__ __forceinline__ float u01(unsigned int h) {
    return fmaxf((h >> 8) * (1.f / 16777216.f), 1e-7f);
}

__global__ __launch_bounds__(256) void k_query(
        const int*   __restrict__ occ_idx,
        const int*   __restrict__ cum,
        const float* __restrict__ params,
        int*         __restrict__ qi_out,
        float*       __restrict__ off_out) {
    int q = blockIdx.x * blockDim.x + threadIdx.x;
    if (q >= QMAX) return;

    int total = cum[NNZ - 1];
    bool valid = q < total;

    // searchsorted(cum, q, side='right')
    int lo = 0, hi = NNZ;
    while (lo < hi) {
        int mid = (lo + hi) >> 1;
        if (cum[mid] <= q) lo = mid + 1; else hi = mid;
    }
    int src = lo < (NNZ - 1) ? lo : (NNZ - 1);

    int b_ = occ_idx[src];
    int i_ = occ_idx[NNZ + src];
    int j_ = occ_idx[2 * NNZ + src];

    float xa = params[(size_t)src * 4 + 0];
    float xb = params[(size_t)src * 4 + 1];
    float ya = params[(size_t)src * 4 + 2];
    float yb = params[(size_t)src * 4 + 3];

    unsigned int s0 = (unsigned int)q * 2654435761u;
    float u1 = u01(pcg_hash(s0 + 0x9e3779b9u));
    float u2 = u01(pcg_hash(s0 + 0x7f4a7c15u));
    float u3 = u01(pcg_hash(s0 + 0x94d049bbu));
    float u4 = u01(pcg_hash(s0 + 0xbf58476du));

    float px = __powf(u1, 1.f / xa);
    float py = __powf(u2, 1.f / xb);
    float x  = px / (px + py + 1e-20f);
    float qx = __powf(u3, 1.f / ya);
    float qy = __powf(u4, 1.f / yb);
    float y  = qx / (qx + qy + 1e-20f);

    qi_out[(size_t)q * 3 + 0] = valid ? b_ : 0;
    qi_out[(size_t)q * 3 + 1] = valid ? i_ : 0;
    qi_out[(size_t)q * 3 + 2] = valid ? j_ : 0;
    off_out[(size_t)q * 2 + 0] = valid ? x : 0.f;
    off_out[(size_t)q * 2 + 1] = valid ? y : 0.f;
}

// ---------------------------------------------------------------------
extern "C" void kernel_launch(void* const* d_in, const int* in_sizes, int n_in,
                              void* d_out, int out_size, void* d_ws, size_t ws_size,
                              hipStream_t stream) {
    const float* occ_values = (const float*)d_in[0];   // [NNZ, 16]
    const int*   occ_idx    = (const int*)  d_in[1];   // [3, NNZ]
    const float* pixel      = (const float*)d_in[2];   // [NNZ, 128]
    const float* w1         = (const float*)d_in[3];   // [3,3,128,128]
    const float* b1         = (const float*)d_in[4];   // [128]
    const float* w2         = (const float*)d_in[5];   // [128,4]
    const float* b2         = (const float*)d_in[6];   // [4]

    // workspace layout (all segments 16-byte aligned)
    int*      counts  = (int*)d_ws;
    int*      cum     = counts + NNZ;                       //  +400000 B
    int*      grid    = cum + NNZ;                          //  +400000 B
    float*    params  = (float*)(grid + GRIDN);             //  +8 MB
    _Float16* w1h     = (_Float16*)(params + (size_t)NNZ * 4);   // +1.6 MB
    _Float16* pixel_h = w1h + 9 * CIN * CHID;               //  +288 KB

    int*   qi_out  = (int*)d_out;
    float* off_out = (float*)d_out + (size_t)QMAX * 3;

    k_counts<<<(NNZ + 255) / 256, 256, 0, stream>>>(occ_values, counts);
    k_scan<<<1, 1024, 0, stream>>>(counts, cum);
    k_grid_init<<<(GRIDN + 255) / 256, 256, 0, stream>>>(grid);
    k_grid_scatter<<<(NNZ + 255) / 256, 256, 0, stream>>>(occ_idx, grid);
    k_w1_to_f16<<<(9 * CIN * CHID + 255) / 256, 256, 0, stream>>>(w1, w1h);
    k_pix_to_f16<<<((NNZ * CIN / 4) + 255) / 256, 256, 0, stream>>>(pixel, pixel_h);
    k_subm_conv<<<(NNZ + BM - 1) / BM, 256, 0, stream>>>(
        occ_idx, pixel_h, w1h, b1, w2, b2, grid, params);
    k_query<<<(QMAX + 255) / 256, 256, 0, stream>>>(
        occ_idx, cum, params, qi_out, off_out);
}
